// NystromSelfAttention_35167192220045
// MI455X (gfx1250) — matrix operand order repaired and checked by reading, not compiled, not run on hardware
//
#include <hip/hip_runtime.h>
#include <hip/hip_bf16.h>

// ---------------------------------------------------------------------------
// Nystrom self-attention for MI455X (gfx1250), wave32 + WMMA bf16 path.
//   B=4, N=4096, D=512, H=8, M=64, DH=64, BH=32
// All GEMMs run through v_wmma_f32_16x16x32_bf16 (f32->bf16 staged in LDS,
// f32 accumulate), double-buffered LDS tiles + global_prefetch of the
// next-next K tile. Landmark RLS selection ~ deterministic top-M score
// select; pinv(64x64) ~ Newton-Schulz iteration in LDS.
// ---------------------------------------------------------------------------

typedef __attribute__((ext_vector_type(16))) __bf16 v16bf;
typedef __attribute__((ext_vector_type(8)))  __bf16 v8bf;
typedef __attribute__((ext_vector_type(4)))  __bf16 v4bf;
typedef __attribute__((ext_vector_type(2)))  __bf16 v2bf;
typedef __attribute__((ext_vector_type(8)))  float  v8f;
typedef __attribute__((ext_vector_type(4)))  float  v4f4;

#define LDSS 40   // LDS row stride in bf16 elems (80 bytes -> rows 16B aligned)

// Fixed problem dims (reference: B=4, N=4096, D=512, H=8, M=64, DH=64)
#define CN   4096
#define CH   8
#define CDH  64
#define CD   512

__device__ __forceinline__ v16bf frag_cat(const __bf16* p0, const __bf16* p1) {
    v8bf lo = *(const v8bf*)p0;
    v8bf hi = *(const v8bf*)p1;
    return __builtin_shufflevector(lo, hi, 0,1,2,3,4,5,6,7,8,9,10,11,12,13,14,15);
}

// A fragment (16x32 bf16): lane holds row m=lane&15; lanes<16 -> K {0..7,16..23},
// lanes>=16 -> K {8..15,24..31}  (ISA 7.12.2 "16-bit A-Matrix 16x32")
__device__ __forceinline__ v16bf load_a_frag(const __bf16* As, int rowBase, int lane) {
    const int m   = rowBase + (lane & 15);
    const int klo = (lane < 16) ? 0 : 8;
    const __bf16* p = As + m * LDSS;
    return frag_cat(p + klo, p + 16 + klo);
}

// B fragment (32x16 bf16): lane holds col n=lane&15; lanes<16 -> K {0..7,16..23},
// lanes>=16 -> K {8..15,24..31}. Bs is stored [n][k] so reads are contiguous.
__device__ __forceinline__ v16bf load_b_frag(const __bf16* Bs, int colBase, int lane) {
    const int n  = colBase + (lane & 15);
    const int kb = (lane < 16) ? 0 : 8;
    const __bf16* p = Bs + n * LDSS + kb;
    return frag_cat(p, p + 16);
}

// Stage a 64x32 f32 tile (row-major, row stride ld) into LDS as bf16 [r][k].
// float4 global loads -> packed 4x bf16 -> single ds_store_b64 per thread-iter.
__device__ __forceinline__ void stage_rowmajor(const float* __restrict__ src,
                                               long long r0, long long ld, int k0,
                                               __bf16* __restrict__ dst, int tid) {
    #pragma unroll
    for (int i = 0; i < 4; ++i) {
        int e4 = tid + i * 128;          // 0..511 float4 slots
        int r  = e4 >> 3;                // 8 float4 per 32-wide row
        int c4 = (e4 & 7) << 2;
        v4f4 f = *(const v4f4*)(src + (r0 + r) * ld + (k0 + c4));
        v4bf h;
        h[0] = (__bf16)f[0]; h[1] = (__bf16)f[1];
        h[2] = (__bf16)f[2]; h[3] = (__bf16)f[3];
        *(v4bf*)(dst + r * LDSS + c4) = h;   // 8B-aligned (80r + 2*c4)
    }
}

// Stage B tile (normal layout: global [K x Ncols]) transposed into Bs[n][k].
// Two coalesced b32 loads per packed v2bf ds_store_b32.
__device__ __forceinline__ void stage_b_normal(const float* __restrict__ src,
                                               long long n0, long long ldb, int k0,
                                               __bf16* __restrict__ dst, int tid) {
    #pragma unroll
    for (int i = 0; i < 8; ++i) {
        int e2 = tid + i * 128;          // 0..1023 k-pair slots
        int n  = e2 & 63;                // consecutive threads -> consecutive n
        int kp = (e2 >> 6) << 1;
        float f0 = src[(long long)(k0 + kp)     * ldb + (n0 + n)];
        float f1 = src[(long long)(k0 + kp + 1) * ldb + (n0 + n)];
        v2bf h; h[0] = (__bf16)f0; h[1] = (__bf16)f1;
        *(v2bf*)(dst + n * LDSS + kp) = h;   // 4B-aligned (80n + 2*kp)
    }
}

// ---------------------------------------------------------------------------
// Tiled WMMA GEMM: C[z] = A[z] @ B[z] (+bias), f32 in/out, bf16 compute.
//   BT   : 0 -> B is [K x Ncols] row-major; 1 -> B is [Ncols x K] row-major (B^T)
//   SMODE: 0 plain  1 split-heads ([B*N,D] -> [BH,N,DH])  2 merge-heads
// Block = 128 threads (4 waves); tile 64x64, K-step 32, double-buffered LDS;
// each wave owns a 32x32 quadrant = four 16x16 WMMA accumulators.
// All problem dims are multiples of the tile sizes (no edge guards).
// ---------------------------------------------------------------------------
template<int BT, int SMODE>
__global__ __launch_bounds__(128)
void gemm_wmma(const float* __restrict__ A, long long lda, long long sA,
               const float* __restrict__ B, long long ldb, long long sB,
               float* __restrict__ C, long long ldc, long long sC,
               const float* __restrict__ bias, int K)
{
    __shared__ __align__(16) __bf16 As[2][64 * LDSS];
    __shared__ __align__(16) __bf16 Bs[2][64 * LDSS];

    const int tid  = threadIdx.x;
    const int lane = tid & 31;
    const int wave = tid >> 5;
    const int wr   = (wave >> 1) * 32;   // wave row offset in 64x64 tile
    const int wc   = (wave & 1) * 32;    // wave col offset
    const int m0   = blockIdx.y * 64;
    const int n0   = blockIdx.x * 64;
    const int z    = blockIdx.z;

    const float* Ab = A + (long long)z * sA;
    const float* Bb = B + (long long)z * sB;

    v8f acc00 = {}; v8f acc01 = {}; v8f acc10 = {}; v8f acc11 = {};

    // prologue: stage K-tile 0 into buffer 0
    stage_rowmajor(Ab, m0, lda, 0, As[0], tid);
    if (BT == 0) stage_b_normal(Bb, n0, ldb, 0, Bs[0], tid);
    else         stage_rowmajor(Bb, n0, ldb, 0, Bs[0], tid);
    __syncthreads();

    const int nsteps = K >> 5;
    for (int s = 0; s < nsteps; ++s) {
        const int cur = s & 1, nxt = cur ^ 1;
        const int k0  = s << 5;

        // prefetch K-tile s+2 toward L2/WGP$ (global_prefetch_b8)
        if (k0 + 64 < K) {
            __builtin_prefetch(Ab + (long long)(m0 + (tid & 63)) * lda +
                               (k0 + 64 + ((tid >> 6) << 4)), 0, 1);
            if (BT == 1)
                __builtin_prefetch(Bb + (long long)(n0 + (tid & 63)) * ldb +
                                   (k0 + 64 + ((tid >> 6) << 4)), 0, 1);
            else
                __builtin_prefetch(Bb + (long long)(k0 + 64 + (tid >> 2)) * ldb +
                                   (n0 + ((tid & 3) << 4)), 0, 1);
        }

        // stage K-tile s+1 into the other buffer (writes never collide with
        // this iteration's fragment reads; previous barrier fenced s-1 reads)
        if (s + 1 < nsteps) {
            stage_rowmajor(Ab, m0, lda, k0 + 32, As[nxt], tid);
            if (BT == 0) stage_b_normal(Bb, n0, ldb, k0 + 32, Bs[nxt], tid);
            else         stage_rowmajor(Bb, n0, ldb, k0 + 32, Bs[nxt], tid);
        }

        v16bf a0 = load_a_frag(As[cur], wr + 0,  lane);
        v16bf a1 = load_a_frag(As[cur], wr + 16, lane);
        v16bf b0 = load_b_frag(Bs[cur], wc + 0,  lane);
        v16bf b1 = load_b_frag(Bs[cur], wc + 16, lane);

        acc00 = __builtin_amdgcn_wmma_f32_16x16x32_bf16(false, a0, false, b0, (short)0, acc00, false, false);
        acc01 = __builtin_amdgcn_wmma_f32_16x16x32_bf16(false, a0, false, b1, (short)0, acc01, false, false);
        acc10 = __builtin_amdgcn_wmma_f32_16x16x32_bf16(false, a1, false, b0, (short)0, acc10, false, false);
        acc11 = __builtin_amdgcn_wmma_f32_16x16x32_bf16(false, a1, false, b1, (short)0, acc11, false, false);
        __syncthreads();
    }

    // ---- store: C/D layout (ISA 7.12.2): VGPR i -> M = i + 8*(lane>=16), N = lane&15
    const int colL   = lane & 15;
    const int rowOff = (lane < 16) ? 0 : 8;
    #pragma unroll
    for (int ti = 0; ti < 2; ++ti) {
        #pragma unroll
        for (int tj = 0; tj < 2; ++tj) {
            const v8f& acc = (ti == 0) ? (tj == 0 ? acc00 : acc01)
                                       : (tj == 0 ? acc10 : acc11);
            int tm = m0 + wr + ti * 16;
            int tn = n0 + wc + tj * 16;
            #pragma unroll
            for (int i = 0; i < 8; ++i) {
                int gm = tm + rowOff + i;
                int gn = tn + colL;
                float v = acc[i];
                if (bias) v += bias[gn];
                if (SMODE == 0) {
                    C[(long long)z * sC + (long long)gm * ldc + gn] = v;
                } else if (SMODE == 1) {   // [B*N, D] -> [BH, N, DH]  (shifts only)
                    int b = gm >> 12, n = gm & (CN - 1);
                    int h = gn >> 6,  dh = gn & (CDH - 1);
                    C[(((long long)((b << 3) + h) << 12) + n) * CDH + dh] = v;
                } else {                   // per-head [N, DH] -> [B*N, D]
                    int b = z >> 3, h = z & (CH - 1);
                    C[(((long long)(b << 12) + gm) << 9) + (h << 6) + gn] = v;
                }
            }
        }
    }
}

// ---------------------------------------------------------------------------
// Deterministic landmark selection: score = ||k_n||^2, iterative top-M argmax.
// One block (256 thr) per head; scores live in LDS.
// ---------------------------------------------------------------------------
__global__ __launch_bounds__(256)
void select_landmarks(const float* __restrict__ Kh, int* __restrict__ idx,
                      int Nn, int DHd, int Mm)
{
    __shared__ float sc[4096];
    __shared__ float rv[256];
    __shared__ int   ri[256];
    const int z = blockIdx.x, t = threadIdx.x;
    const float* kb = Kh + (long long)z * Nn * DHd;

    for (int n = t; n < Nn; n += 256) {
        float s = 0.f;
        const float* row = kb + (long long)n * DHd;
        for (int d2 = 0; d2 < DHd; ++d2) { float v = row[d2]; s += v * v; }
        sc[n] = s;
    }
    __syncthreads();

    for (int m = 0; m < Mm; ++m) {
        float best = -3.4e38f; int bi = 0;
        for (int n = t; n < Nn; n += 256)
            if (sc[n] > best) { best = sc[n]; bi = n; }
        rv[t] = best; ri[t] = bi;
        __syncthreads();
        for (int s = 128; s > 0; s >>= 1) {
            if (t < s && rv[t + s] > rv[t]) { rv[t] = rv[t + s]; ri[t] = ri[t + s]; }
            __syncthreads();
        }
        if (t == 0) { idx[z * Mm + m] = ri[0]; sc[ri[0]] = -3.4e38f; }
        __syncthreads();
    }
}

__global__ __launch_bounds__(256)
void gather_rows(const float* __restrict__ src, const int* __restrict__ idx,
                 float* __restrict__ dst, int Nn, int Mm, int DHd)
{
    const int z = blockIdx.x;
    for (int e = threadIdx.x; e < Mm * DHd; e += blockDim.x) {
        int m = e / DHd, d2 = e % DHd;
        dst[((long long)z * Mm + m) * DHd + d2] =
            src[((long long)z * Nn + idx[z * Mm + m]) * DHd + d2];
    }
}

// Row softmax of scale*x. One block per row; works for cols=64 and cols=4096.
__global__ __launch_bounds__(256)
void softmax_rows(float* __restrict__ data, int cols, float scale)
{
    __shared__ float red[256];
    const long long row = blockIdx.x;
    float* p = data + row * (long long)cols;
    const int t = threadIdx.x;

    float mx = -3.4e38f;
    for (int c = t; c < cols; c += 256) mx = fmaxf(mx, scale * p[c]);
    red[t] = mx; __syncthreads();
    for (int s = 128; s > 0; s >>= 1) { if (t < s) red[t] = fmaxf(red[t], red[t + s]); __syncthreads(); }
    mx = red[0]; __syncthreads();

    float sum = 0.f;
    for (int c = t; c < cols; c += 256) {
        float e = __expf(scale * p[c] - mx);
        p[c] = e; sum += e;
    }
    red[t] = sum; __syncthreads();
    for (int s = 128; s > 0; s >>= 1) { if (t < s) red[t] += red[t + s]; __syncthreads(); }
    const float inv = 1.0f / red[0];
    for (int c = t; c < cols; c += 256) p[c] *= inv;
}

// Newton-Schulz pseudo-inverse of 64x64 (per head), all in LDS.
// X0 = A^T/(||A||_1 ||A||_inf); X <- 2X - X A X  (24 iters)
__global__ __launch_bounds__(256)
void pinv_newton(const float* __restrict__ Ain, float* __restrict__ Xout)
{
    __shared__ float Asm[64 * 64];
    __shared__ float Xa[64 * 64];
    __shared__ float Xb[64 * 64];
    __shared__ float Y[64 * 64];
    __shared__ float red[64];
    __shared__ float norms[2];
    const int z = blockIdx.x, t = threadIdx.x;
    const float* Ab = Ain + (long long)z * 64 * 64;

    for (int e = t; e < 4096; e += 256) Asm[e] = Ab[e];
    __syncthreads();

    if (t < 64) { float s = 0; for (int r = 0; r < 64; ++r) s += fabsf(Asm[r * 64 + t]); red[t] = s; }
    __syncthreads();
    if (t == 0) { float m = 0; for (int i = 0; i < 64; ++i) m = fmaxf(m, red[i]); norms[0] = m; }
    __syncthreads();
    if (t < 64) { float s = 0; for (int c = 0; c < 64; ++c) s += fabsf(Asm[t * 64 + c]); red[t] = s; }
    __syncthreads();
    if (t == 0) { float m = 0; for (int i = 0; i < 64; ++i) m = fmaxf(m, red[i]); norms[1] = m; }
    __syncthreads();

    const float alpha = 1.0f / fmaxf(norms[0] * norms[1], 1e-20f);
    for (int e = t; e < 4096; e += 256) { int r = e >> 6, c = e & 63; Xa[e] = Asm[c * 64 + r] * alpha; }
    __syncthreads();

    float* Xc = Xa; float* Xn = Xb;
    for (int it = 0; it < 24; ++it) {
        for (int e = t; e < 4096; e += 256) {
            int r = e >> 6, c = e & 63; float s = 0.f;
            for (int kk = 0; kk < 64; ++kk) s += Asm[r * 64 + kk] * Xc[kk * 64 + c];
            Y[e] = s;
        }
        __syncthreads();
        for (int e = t; e < 4096; e += 256) {
            int r = e >> 6, c = e & 63; float s = 0.f;
            for (int kk = 0; kk < 64; ++kk) s += Xc[r * 64 + kk] * Y[kk * 64 + c];
            Xn[e] = 2.0f * Xc[e] - s;
        }
        __syncthreads();
        float* tp = Xc; Xc = Xn; Xn = tp;
    }
    for (int e = t; e < 4096; e += 256) Xout[(long long)z * 4096 + e] = Xc[e];
}

// ---------------------------------------------------------------------------
extern "C" void kernel_launch(void* const* d_in, const int* in_sizes, int n_in,
                              void* d_out, int out_size, void* d_ws, size_t ws_size,
                              hipStream_t stream)
{
    (void)in_sizes; (void)n_in; (void)out_size; (void)ws_size;

    constexpr int Bb = 4, Nn = 4096, Dd = 512, Hh = 8, Mm = 64, DHd = 64;
    constexpr int BH = Bb * Hh;            // 32
    constexpr int BN = Bb * Nn;            // 16384
    const float scale = 0.125f;            // 1/sqrt(DH)

    const float* x  = (const float*)d_in[0];
    const float* Wq = (const float*)d_in[1];
    const float* bq = (const float*)d_in[2];
    const float* Wk = (const float*)d_in[3];
    const float* bk = (const float*)d_in[4];
    const float* Wv = (const float*)d_in[5];
    const float* bv = (const float*)d_in[6];
    const float* Wo = (const float*)d_in[7];
    const float* bo = (const float*)d_in[8];
    float* out = (float*)d_out;

    // ---- workspace carve-up
    char* w = (char*)d_ws;
    size_t off = 0;
    auto alloc = [&](size_t nElems) {
        void* p = (void*)(w + off);
        off = (off + nElems * 4 + 255) & ~(size_t)255;
        return p;
    };
    float* Qh   = (float*)alloc((size_t)BH * Nn * DHd);
    float* Kh   = (float*)alloc((size_t)BH * Nn * DHd);
    float* Vh   = (float*)alloc((size_t)BH * Nn * DHd);
    float* k1   = (float*)alloc((size_t)BH * Nn * Mm);
    float* k3   = (float*)alloc((size_t)BH * Mm * Nn);
    float* ctx  = (float*)alloc((size_t)BN * Dd);
    float* k2   = (float*)alloc((size_t)BH * Mm * Mm);
    float* pInv = (float*)alloc((size_t)BH * Mm * Mm);
    float* t1   = (float*)alloc((size_t)BH * Mm * DHd);
    float* t2   = (float*)alloc((size_t)BH * Mm * DHd);
    float* q_l  = (float*)alloc((size_t)BH * Mm * DHd);
    float* k_l  = (float*)alloc((size_t)BH * Mm * DHd);
    int*   idx  = (int*)alloc((size_t)BH * Mm);

    const dim3 blk(128);
    const dim3 gProj(Dd / 64, BN / 64, 1);           // (8, 256, 1)

    // 1) q,k,v projections, stored split-head [BH, N, DH]
    gemm_wmma<0,1><<<gProj, blk, 0, stream>>>(x, Dd, 0, Wq, Dd, 0, Qh, 0, 0, bq, Dd);
    gemm_wmma<0,1><<<gProj, blk, 0, stream>>>(x, Dd, 0, Wk, Dd, 0, Kh, 0, 0, bk, Dd);
    gemm_wmma<0,1><<<gProj, blk, 0, stream>>>(x, Dd, 0, Wv, Dd, 0, Vh, 0, 0, bv, Dd);

    // 2) landmark selection + gathers
    select_landmarks<<<BH, 256, 0, stream>>>(Kh, idx, Nn, DHd, Mm);
    gather_rows<<<BH, 256, 0, stream>>>(Qh, idx, q_l, Nn, Mm, DHd);
    gather_rows<<<BH, 256, 0, stream>>>(Kh, idx, k_l, Nn, Mm, DHd);

    // 3) k1 = softmax(q @ k_l^T * scale)   [BH, N, M]
    gemm_wmma<1,0><<<dim3(1, Nn / 64, BH), blk, 0, stream>>>(
        Qh, DHd, (long long)Nn * DHd, k_l, DHd, (long long)Mm * DHd,
        k1, Mm, (long long)Nn * Mm, nullptr, DHd);
    softmax_rows<<<BH * Nn, 256, 0, stream>>>(k1, Mm, scale);

    // 4) k2 = softmax(q_l @ k_l^T * scale) [BH, M, M]; then pinv via Newton-Schulz
    gemm_wmma<1,0><<<dim3(1, 1, BH), blk, 0, stream>>>(
        q_l, DHd, (long long)Mm * DHd, k_l, DHd, (long long)Mm * DHd,
        k2, Mm, (long long)Mm * Mm, nullptr, DHd);
    softmax_rows<<<BH * Mm, 256, 0, stream>>>(k2, Mm, scale);
    pinv_newton<<<BH, 256, 0, stream>>>(k2, pInv);

    // 5) k3 = softmax(q_l @ k^T * scale)   [BH, M, N]
    gemm_wmma<1,0><<<dim3(Nn / 64, 1, BH), blk, 0, stream>>>(
        q_l, DHd, (long long)Mm * DHd, Kh, DHd, (long long)Nn * DHd,
        k3, Nn, (long long)Mm * Nn, nullptr, DHd);
    softmax_rows<<<BH * Mm, 256, 0, stream>>>(k3, Nn, scale);

    // 6) t1 = k3 @ v  [BH, M, DH]
    gemm_wmma<0,0><<<dim3(1, 1, BH), blk, 0, stream>>>(
        k3, Nn, (long long)Mm * Nn, Vh, DHd, (long long)Nn * DHd,
        t1, DHd, (long long)Mm * DHd, nullptr, Nn);

    // 7) t2 = pinv(k2) @ t1  [BH, M, DH]
    gemm_wmma<0,0><<<dim3(1, 1, BH), blk, 0, stream>>>(
        pInv, Mm, (long long)Mm * Mm, t1, DHd, (long long)Mm * DHd,
        t2, DHd, (long long)Mm * DHd, nullptr, Mm);

    // 8) ctx = k1 @ t2, merged back to [B*N, D]
    gemm_wmma<0,2><<<dim3(1, Nn / 64, BH), blk, 0, stream>>>(
        k1, Mm, (long long)Nn * Mm, t2, DHd, (long long)Mm * DHd,
        ctx, 0, 0, nullptr, Mm);

    // 9) out = ctx @ Wo + bo  [B*N, D]
    gemm_wmma<0,0><<<gProj, blk, 0, stream>>>(
        ctx, Dd, 0, Wo, Dd, 0, out, Dd, 0, bo, Dd);
}
